// MultiHeadAttention_2104533975673
// MI455X (gfx1250) — matrix operand order repaired and checked by reading, not compiled, not run on hardware
//
#include <hip/hip_runtime.h>

// ---------------------------------------------------------------------------
// MultiHeadAttention for MI455X (gfx1250): bf16 WMMA 16x16x32, f32 accumulate.
// B=2 S=2048 D=1024 H=16 HD=64
// ---------------------------------------------------------------------------

typedef __bf16 bf16;
typedef __attribute__((ext_vector_type(8)))  bf16  bf16x8;
typedef __attribute__((ext_vector_type(16))) bf16  bf16x16;
typedef __attribute__((ext_vector_type(4)))  float f32x4;
typedef __attribute__((ext_vector_type(8)))  float v8f;

static constexpr int Bb  = 2;
static constexpr int Ss  = 2048;
static constexpr int Dd  = 1024;
static constexpr int Hh  = 16;
static constexpr int HDd = 64;
static constexpr float LOG2E = 1.44269504f;
static constexpr float NEGBIG = -3.0e38f;

static __device__ __forceinline__ v8f wmma_bf16(bf16x16 a, bf16x16 b, v8f c) {
  return __builtin_amdgcn_wmma_f32_16x16x32_bf16(
      /*neg_a=*/false, a, /*neg_b=*/false, b,
      /*c_mod=*/(short)0, c, /*reuse_a=*/false, /*reuse_b=*/false);
}

// ---------------------------------------------------------------------------
// Kernel 0: transpose + convert the 4 weight matrices: Wt[m][n][k] = bf16(W[k][n])
// ---------------------------------------------------------------------------
__global__ void __launch_bounds__(256)
mha_pack_w(const float* __restrict__ Wq, const float* __restrict__ Wk,
           const float* __restrict__ Wv, const float* __restrict__ Wo,
           bf16* __restrict__ wt) {
  size_t tid = (size_t)blockIdx.x * 256 + threadIdx.x;   // 4*D*D threads
  int    m   = (int)(tid / ((size_t)Dd * Dd));
  size_t r   = tid % ((size_t)Dd * Dd);
  int    k   = (int)(r / Dd);
  int    n   = (int)(r % Dd);
  const float* W = (m == 0) ? Wq : (m == 1) ? Wk : (m == 2) ? Wv : Wo;
  wt[(size_t)m * Dd * Dd + (size_t)n * Dd + k] = (bf16)W[r];  // W[k*D+n], coalesced read
}

// ---------------------------------------------------------------------------
// Kernel 1: QKV projection. One wave -> 16x64 tile (one head's HD slice).
// Q,K stored [b][h][s][hd] bf16; V stored transposed [b][h][hd][s] bf16.
// grid(32, H, 3), block(32, 8)
// ---------------------------------------------------------------------------
__global__ void __launch_bounds__(256)
mha_qkv_gemm(const float* __restrict__ Xq, const float* __restrict__ Xk,
             const float* __restrict__ Xv,
             const float* __restrict__ bq, const float* __restrict__ bk,
             const float* __restrict__ bv,
             const bf16* __restrict__ wt,
             bf16* __restrict__ Qo, bf16* __restrict__ Ko, bf16* __restrict__ Vt) {
  const int lane = threadIdx.x;
  const int l16  = lane & 15;
  const int hi   = lane >> 4;
  const int rowtile = blockIdx.x * 8 + threadIdx.y;   // 0..255
  const int row0    = rowtile * 16;                   // flat row over B*S
  const int head    = blockIdx.y;
  const int mat     = blockIdx.z;

  const float* X    = (mat == 0) ? Xq : (mat == 1) ? Xk : Xv;
  const float* bias = (mat == 0) ? bq : (mat == 1) ? bk : bv;
  const bf16*  W    = wt + (size_t)mat * Dd * Dd;     // Wt[n][k] layout
  const int colbase = head * HDd;

  v8f acc[4] = {};
  const float* arow = X + (size_t)(row0 + l16) * Dd;

  for (int kb = 0; kb < Dd; kb += 32) {
    // A fragment (16x32 bf16): lane (m=l16, hi) holds K = hi*16..hi*16+15
    f32x4 f0 = *(const f32x4*)(arow + kb + hi * 16);
    f32x4 f1 = *(const f32x4*)(arow + kb + hi * 16 + 4);
    f32x4 f2 = *(const f32x4*)(arow + kb + hi * 16 + 8);
    f32x4 f3 = *(const f32x4*)(arow + kb + hi * 16 + 12);
    bf16x16 a;
#pragma unroll
    for (int e = 0; e < 4; ++e) {
      a[e]      = (bf16)f0[e];
      a[4 + e]  = (bf16)f1[e];
      a[8 + e]  = (bf16)f2[e];
      a[12 + e] = (bf16)f3[e];
    }
#pragma unroll
    for (int nt = 0; nt < 4; ++nt) {
      // B fragment (32x16): lane (n=l16, hi) holds K = hi*16..hi*16+15, contiguous in Wt
      const bf16* wrow = W + (size_t)(colbase + nt * 16 + l16) * Dd + kb + hi * 16;
      bf16x16 bfrag = *(const bf16x16*)wrow;
      acc[nt] = wmma_bf16(a, bfrag, acc[nt]);
    }
  }

  const int b  = row0 >> 11;           // / S
  const int s0 = row0 & (Ss - 1);
#pragma unroll
  for (int nt = 0; nt < 4; ++nt) {
    const int hd  = nt * 16 + l16;
    const float bval = bias[colbase + hd];
#pragma unroll
    for (int r = 0; r < 8; ++r) {
      const int s = s0 + hi * 8 + r;   // C/D row = hi*8+r
      const float v = acc[nt][r] + bval;
      if (mat == 0)
        Qo[(((size_t)(b * Hh + head)) * Ss + s) * HDd + hd] = (bf16)v;
      else if (mat == 1)
        Ko[(((size_t)(b * Hh + head)) * Ss + s) * HDd + hd] = (bf16)v;
      else
        Vt[(((size_t)(b * Hh + head)) * HDd + hd) * Ss + s] = (bf16)v;
    }
  }
}

// ---------------------------------------------------------------------------
// Kernel 2: flash attention. One wave per (b,h,16-query tile); 32 keys/iter.
// grid(512), block(32,8); 8 KB LDS for the P bounce.
// ---------------------------------------------------------------------------
__global__ void __launch_bounds__(256)
mha_attn(const bf16* __restrict__ Q, const bf16* __restrict__ K,
         const bf16* __restrict__ V, const int* __restrict__ pad,
         bf16* __restrict__ ctx) {
  __shared__ bf16 plds[8 * 16 * 32];

  const int lane = threadIdx.x;
  const int l16  = lane & 15;
  const int hi   = lane >> 4;
  const int w    = threadIdx.y;
  const int t    = blockIdx.x * 8 + w;     // 0..4095
  const int b    = t >> 11;                // H*(S/16) = 2048 tiles/batch
  const int h    = (t >> 7) & 15;          // 128 tiles/head
  const int qbase = (t & 127) * 16;

  const bf16* Qh = Q + (size_t)(b * Hh + h) * Ss * HDd;
  const bf16* Kh = K + (size_t)(b * Hh + h) * Ss * HDd;
  const bf16* Vh = V + (size_t)(b * Hh + h) * HDd * Ss;   // transposed [hd][s]
  const int* padb = pad + b * Ss;

  // Q A-fragments: 16x64 = two 16x32 fragments
  const bf16* qrow = Qh + (size_t)(qbase + l16) * HDd;
  bf16x16 aQ[2];
#pragma unroll
  for (int kk = 0; kk < 2; ++kk) {
    bf16x8 lo  = *(const bf16x8*)(qrow + kk * 32 + hi * 16);
    bf16x8 hi8 = *(const bf16x8*)(qrow + kk * 32 + hi * 16 + 8);
    aQ[kk] = __builtin_shufflevector(lo, hi8, 0, 1, 2, 3, 4, 5, 6, 7,
                                     8, 9, 10, 11, 12, 13, 14, 15);
  }

  float rm[8], rl[8];
  v8f o[4] = {};
#pragma unroll
  for (int r = 0; r < 8; ++r) { rm[r] = NEGBIG; rl[r] = 0.f; }

  bf16* pl = plds + w * (16 * 32);
  const int nblocks = (qbase + 16 + 31) >> 5;   // causal key-block count

  for (int blk = 0; blk < nblocks; ++blk) {
    const int kb = blk * 32;

    // ---- scores: S = Q(16x64) . K^T(64x32), two 16-key halves -------------
    v8f sc[2];
#pragma unroll
    for (int hh = 0; hh < 2; ++hh) {
      const bf16* krow = Kh + (size_t)(kb + hh * 16 + l16) * HDd;
      bf16x16 b0 = *(const bf16x16*)(krow + hi * 16);        // hd 0..31
      bf16x16 b1 = *(const bf16x16*)(krow + 32 + hi * 16);   // hd 32..63
      v8f c = {};
      c = wmma_bf16(aQ[0], b0, c);
      c = wmma_bf16(aQ[1], b1, c);
      sc[hh] = c;
    }

    // ---- scale + masks + online softmax -----------------------------------
    const int k0i = kb + l16, k1i = kb + 16 + l16;
    const int pv0 = padb[k0i], pv1 = padb[k1i];
    float bmax[8], corr[8], psum[8];
#pragma unroll
    for (int r = 0; r < 8; ++r) {
      const int qg = qbase + hi * 8 + r;
      float s0 = sc[0][r] * 0.125f;                 // 1/sqrt(64)
      float s1 = sc[1][r] * 0.125f;
      if (k0i > qg || pv0) s0 = NEGBIG;
      if (k1i > qg || pv1) s1 = NEGBIG;
      sc[0][r] = s0; sc[1][r] = s1;
      bmax[r] = fmaxf(s0, s1);
    }
#pragma unroll
    for (int off = 1; off < 16; off <<= 1)
#pragma unroll
      for (int r = 0; r < 8; ++r)
        bmax[r] = fmaxf(bmax[r], __shfl_xor(bmax[r], off, 16));

#pragma unroll
    for (int r = 0; r < 8; ++r) {
      const float mnew = fmaxf(rm[r], bmax[r]);
      corr[r] = exp2f((rm[r] - mnew) * LOG2E);
      const float p0 = exp2f((sc[0][r] - mnew) * LOG2E);
      const float p1 = exp2f((sc[1][r] - mnew) * LOG2E);
      sc[0][r] = p0; sc[1][r] = p1;
      psum[r] = p0 + p1;
      rm[r] = mnew;
    }
#pragma unroll
    for (int off = 1; off < 16; off <<= 1)
#pragma unroll
      for (int r = 0; r < 8; ++r)
        psum[r] += __shfl_xor(psum[r], off, 16);
#pragma unroll
    for (int r = 0; r < 8; ++r)
      rl[r] = rl[r] * corr[r] + psum[r];
#pragma unroll
    for (int nt = 0; nt < 4; ++nt)
#pragma unroll
      for (int r = 0; r < 8; ++r)
        o[nt][r] *= corr[r];

    // ---- P (C-layout) -> LDS -> A-fragment layout -------------------------
#pragma unroll
    for (int hh = 0; hh < 2; ++hh)
#pragma unroll
      for (int r = 0; r < 8; ++r)
        pl[(hi * 8 + r) * 32 + hh * 16 + l16] = (bf16)sc[hh][r];
    asm volatile("s_wait_dscnt 0x0" ::: "memory");
    bf16x8 plo = *(const bf16x8*)(pl + l16 * 32 + hi * 16);
    bf16x8 phi = *(const bf16x8*)(pl + l16 * 32 + hi * 16 + 8);
    bf16x16 aP = __builtin_shufflevector(plo, phi, 0, 1, 2, 3, 4, 5, 6, 7,
                                         8, 9, 10, 11, 12, 13, 14, 15);

    // ---- O += P(16x32) . V(32x64) ----------------------------------------
#pragma unroll
    for (int nt = 0; nt < 4; ++nt) {
      bf16x16 bv = *(const bf16x16*)(Vh + (size_t)(nt * 16 + l16) * Ss + kb + hi * 16);
      o[nt] = wmma_bf16(aP, bv, o[nt]);
    }
  }

  // ---- normalize + write context [b][s][D] bf16 ---------------------------
#pragma unroll
  for (int nt = 0; nt < 4; ++nt) {
    const int col = h * HDd + nt * 16 + l16;
#pragma unroll
    for (int r = 0; r < 8; ++r) {
      const int srow = qbase + hi * 8 + r;
      const float val = o[nt][r] / rl[r];
      ctx[((size_t)b * Ss + srow) * Dd + col] = (bf16)val;
    }
  }
}

// ---------------------------------------------------------------------------
// Kernel 3: output projection ctx(bf16) @ Wo + bo -> f32. grid(32,16) blk(32,8)
// ---------------------------------------------------------------------------
__global__ void __launch_bounds__(256)
mha_out_gemm(const bf16* __restrict__ ctx, const bf16* __restrict__ wto,
             const float* __restrict__ bo, float* __restrict__ out) {
  const int lane = threadIdx.x;
  const int l16  = lane & 15;
  const int hi   = lane >> 4;
  const int rowtile = blockIdx.x * 8 + threadIdx.y;
  const int row0    = rowtile * 16;
  const int colbase = blockIdx.y * 64;

  v8f acc[4] = {};
  const bf16* arow = ctx + (size_t)(row0 + l16) * Dd;

  for (int kb = 0; kb < Dd; kb += 32) {
    bf16x8 lo  = *(const bf16x8*)(arow + kb + hi * 16);
    bf16x8 hi8 = *(const bf16x8*)(arow + kb + hi * 16 + 8);
    bf16x16 a = __builtin_shufflevector(lo, hi8, 0, 1, 2, 3, 4, 5, 6, 7,
                                        8, 9, 10, 11, 12, 13, 14, 15);
#pragma unroll
    for (int nt = 0; nt < 4; ++nt) {
      const bf16* wrow = wto + (size_t)(colbase + nt * 16 + l16) * Dd + kb + hi * 16;
      bf16x16 bfrag = *(const bf16x16*)wrow;
      acc[nt] = wmma_bf16(a, bfrag, acc[nt]);
    }
  }
#pragma unroll
  for (int nt = 0; nt < 4; ++nt) {
    const int col = colbase + nt * 16 + l16;
    const float bval = bo[col];
#pragma unroll
    for (int r = 0; r < 8; ++r)
      out[(size_t)(row0 + hi * 8 + r) * Dd + col] = acc[nt][r] + bval;
  }
}

// ---------------------------------------------------------------------------
// Launch. Inputs: 0 query, 1 key, 2 value, 3 mask(unused: causal analytic),
// 4 Wq, 5 bq, 6 Wk, 7 bk, 8 Wv, 9 bv, 10 Wo, 11 bo, 12 padding_mask.
// Workspace (~40 MB): Wt(4*D*D) | Q | K | Vt | ctx  (all bf16)
// ---------------------------------------------------------------------------
extern "C" void kernel_launch(void* const* d_in, const int* in_sizes, int n_in,
                              void* d_out, int out_size, void* d_ws, size_t ws_size,
                              hipStream_t stream) {
  (void)in_sizes; (void)n_in; (void)out_size; (void)ws_size;

  const float* q_in = (const float*)d_in[0];
  const float* k_in = (const float*)d_in[1];
  const float* v_in = (const float*)d_in[2];
  const float* Wq   = (const float*)d_in[4];
  const float* bq   = (const float*)d_in[5];
  const float* Wk   = (const float*)d_in[6];
  const float* bk   = (const float*)d_in[7];
  const float* Wv   = (const float*)d_in[8];
  const float* bv   = (const float*)d_in[9];
  const float* Wo   = (const float*)d_in[10];
  const float* bo   = (const float*)d_in[11];
  const int*   pad  = (const int*)d_in[12];
  float* out = (float*)d_out;

  bf16* wt  = (bf16*)d_ws;                       // 4*D*D
  bf16* Qb  = wt + (size_t)4 * Dd * Dd;          // B*S*D
  bf16* Kb  = Qb + (size_t)Bb * Ss * Dd;
  bf16* Vt  = Kb + (size_t)Bb * Ss * Dd;         // transposed per head
  bf16* ctx = Vt + (size_t)Bb * Ss * Dd;

  mha_pack_w<<<dim3((4 * Dd * Dd) / 256), dim3(256), 0, stream>>>(Wq, Wk, Wv, Wo, wt);
  mha_qkv_gemm<<<dim3(32, Hh, 3), dim3(32, 8), 0, stream>>>(
      q_in, k_in, v_in, bq, bk, bv, wt, Qb, Kb, Vt);
  mha_attn<<<dim3(512), dim3(32, 8), 0, stream>>>(Qb, Kb, Vt, pad, ctx);
  mha_out_gemm<<<dim3(32, 16), dim3(32, 8), 0, stream>>>(
      ctx, wt + (size_t)3 * Dd * Dd, bo, out);
}
